// ShapeEncoder_72507637891435
// MI455X (gfx1250) — matrix operand order repaired and checked by reading, not compiled, not run on hardware
//
#include <hip/hip_runtime.h>
#include <hip/hip_bf16.h>

// ---------------- CDNA5 WMMA types ----------------
typedef __attribute__((ext_vector_type(16))) _Float16 v16h;
typedef __attribute__((ext_vector_type(8)))  float    v8f;

#define NNODES 100000
#define NEDGES 1600000
#define NGRAPH 64
#define N64    (NNODES * 64)
#define EPS_MSG 1e-7f

static inline int cdiv(int a, int b) { return (a + b - 1) / b; }

// ---------------- elementwise helpers ----------------
__global__ void k_relu(const float* __restrict__ in, float* __restrict__ out, int n) {
    int t = blockIdx.x * blockDim.x + threadIdx.x;
    if (t < n) out[t] = fmaxf(in[t], 0.f);
}

__global__ void k_add(const float* __restrict__ a, const float* __restrict__ b,
                      float* __restrict__ o, int n) {
    int t = blockIdx.x * blockDim.x + threadIdx.x;
    if (t < n) o[t] = a[t] + b[t];
}

__global__ void k_fill_u32(unsigned int* __restrict__ p, unsigned int v, int n) {
    int t = blockIdx.x * blockDim.x + threadIdx.x;
    if (t < n) p[t] = v;
}

// ---------------- edge kernels, F=64 (float4 vectorized) ----------------
// msg = relu(xin[src]) + eps  (always > 0 -> uint-ordered atomicMax is exact;
// zero-init of mmax reproduces the reference's isolated-node -> 0.0 rule)
__global__ void k_edge_max64(const float4* __restrict__ xin, const int* __restrict__ src,
                             const int* __restrict__ dst, unsigned int* __restrict__ mmax) {
    int t = blockIdx.x * blockDim.x + threadIdx.x;
    if (t >= NEDGES * 16) return;
    int e = t >> 4, c = t & 15;
    int s = src[e], d = dst[e];
    float4 v = xin[s * 16 + c];
    unsigned int* mp = mmax + d * 64 + c * 4;
    atomicMax(mp + 0, __float_as_uint(fmaxf(v.x, 0.f) + EPS_MSG));
    atomicMax(mp + 1, __float_as_uint(fmaxf(v.y, 0.f) + EPS_MSG));
    atomicMax(mp + 2, __float_as_uint(fmaxf(v.z, 0.f) + EPS_MSG));
    atomicMax(mp + 3, __float_as_uint(fmaxf(v.w, 0.f) + EPS_MSG));
}

__global__ void k_edge_sum64(const float4* __restrict__ xin, const int* __restrict__ src,
                             const int* __restrict__ dst, const float4* __restrict__ mmax,
                             float* __restrict__ denom) {
    int t = blockIdx.x * blockDim.x + threadIdx.x;
    if (t >= NEDGES * 16) return;
    int e = t >> 4, c = t & 15;
    int s = src[e], d = dst[e];
    float4 v = xin[s * 16 + c];
    float4 mm = mmax[d * 16 + c];
    float* dp = denom + d * 64 + c * 4;
    atomicAdd(dp + 0, __expf(fmaxf(v.x, 0.f) + EPS_MSG - mm.x));
    atomicAdd(dp + 1, __expf(fmaxf(v.y, 0.f) + EPS_MSG - mm.y));
    atomicAdd(dp + 2, __expf(fmaxf(v.z, 0.f) + EPS_MSG - mm.z));
    atomicAdd(dp + 3, __expf(fmaxf(v.w, 0.f) + EPS_MSG - mm.w));
}

__global__ void k_edge_agg64(const float4* __restrict__ xin, const int* __restrict__ src,
                             const int* __restrict__ dst, const float4* __restrict__ mmax,
                             const float4* __restrict__ denom, float* __restrict__ agg) {
    int t = blockIdx.x * blockDim.x + threadIdx.x;
    if (t >= NEDGES * 16) return;
    int e = t >> 4, c = t & 15;
    int s = src[e], d = dst[e];
    float4 v  = xin[s * 16 + c];
    float4 mm = mmax[d * 16 + c];
    float4 dn = denom[d * 16 + c];
    float* ap = agg + d * 64 + c * 4;
    float m;
    m = fmaxf(v.x, 0.f) + EPS_MSG; atomicAdd(ap + 0, m * __expf(m - mm.x) / (dn.x + 1e-16f));
    m = fmaxf(v.y, 0.f) + EPS_MSG; atomicAdd(ap + 1, m * __expf(m - mm.y) / (dn.y + 1e-16f));
    m = fmaxf(v.z, 0.f) + EPS_MSG; atomicAdd(ap + 2, m * __expf(m - mm.z) / (dn.z + 1e-16f));
    m = fmaxf(v.w, 0.f) + EPS_MSG; atomicAdd(ap + 3, m * __expf(m - mm.w) / (dn.w + 1e-16f));
}

// ---------------- edge kernels, F=6 (cin conv) ----------------
__global__ void k_edge_max6(const float* __restrict__ x, const int* __restrict__ src,
                            const int* __restrict__ dst, unsigned int* __restrict__ mmax) {
    int e = blockIdx.x * blockDim.x + threadIdx.x;
    if (e >= NEDGES) return;
    int s = src[e], d = dst[e];
    #pragma unroll
    for (int j = 0; j < 6; ++j) {
        float m = fmaxf(x[s * 6 + j], 0.f) + EPS_MSG;
        atomicMax(&mmax[d * 6 + j], __float_as_uint(m));
    }
}

__global__ void k_edge_sum6(const float* __restrict__ x, const int* __restrict__ src,
                            const int* __restrict__ dst, const float* __restrict__ mmax,
                            float* __restrict__ denom) {
    int e = blockIdx.x * blockDim.x + threadIdx.x;
    if (e >= NEDGES) return;
    int s = src[e], d = dst[e];
    #pragma unroll
    for (int j = 0; j < 6; ++j) {
        float m = fmaxf(x[s * 6 + j], 0.f) + EPS_MSG;
        atomicAdd(&denom[d * 6 + j], __expf(m - mmax[d * 6 + j]));
    }
}

__global__ void k_edge_agg6(const float* __restrict__ x, const int* __restrict__ src,
                            const int* __restrict__ dst, const float* __restrict__ mmax,
                            const float* __restrict__ denom, float* __restrict__ agg) {
    int e = blockIdx.x * blockDim.x + threadIdx.x;
    if (e >= NEDGES) return;
    int s = src[e], d = dst[e];
    #pragma unroll
    for (int j = 0; j < 6; ++j) {
        float m = fmaxf(x[s * 6 + j], 0.f) + EPS_MSG;
        float ex = __expf(m - mmax[d * 6 + j]);
        atomicAdd(&agg[d * 6 + j], m * ex / (denom[d * 6 + j] + 1e-16f));
    }
}

// ---------------- cin MLP (F too small for WMMA tiles; scalar VALU) ----------------
__global__ void k_cin_mlp1(const float* __restrict__ out0, const float* __restrict__ w1,
                           const float* __restrict__ b1, float* __restrict__ t12,
                           float* __restrict__ stats) {
    __shared__ float ssum[24];
    if (threadIdx.x < 24) ssum[threadIdx.x] = 0.f;
    __syncthreads();
    int i = blockIdx.x * blockDim.x + threadIdx.x;
    if (i < NNODES) {
        float a[6];
        #pragma unroll
        for (int k = 0; k < 6; ++k) a[k] = out0[i * 6 + k];
        #pragma unroll
        for (int j = 0; j < 12; ++j) {
            float v = b1[j];
            #pragma unroll
            for (int k = 0; k < 6; ++k) v += a[k] * w1[k * 12 + j];
            t12[i * 12 + j] = v;
            atomicAdd(&ssum[j], v);          // LDS float atomics (ds_add_f32)
            atomicAdd(&ssum[12 + j], v * v);
        }
    }
    __syncthreads();
    if (threadIdx.x < 24) atomicAdd(&stats[threadIdx.x], ssum[threadIdx.x]);
}

__global__ void k_bn_finalize(const float* __restrict__ stats, const float* __restrict__ g,
                              const float* __restrict__ be, float* __restrict__ scsh,
                              int F, float invN) {
    int j = threadIdx.x;
    if (j >= F) return;
    float mean = stats[j] * invN;
    float var  = stats[F + j] * invN - mean * mean;   // population variance (ddof=0)
    float rs   = rsqrtf(var + 1e-5f);
    float sc   = g[j] * rs;
    scsh[j]     = sc;
    scsh[F + j] = be[j] - mean * sc;
}

__global__ void k_cin_mlp2(const float* __restrict__ t12, const float* __restrict__ scsh,
                           const float* __restrict__ w2, const float* __restrict__ b2,
                           float* __restrict__ h) {
    int t = blockIdx.x * blockDim.x + threadIdx.x;
    if (t >= NNODES * 64) return;
    int i = t >> 6, j = t & 63;
    float acc = b2[j];
    #pragma unroll
    for (int k = 0; k < 12; ++k) {
        float a = fmaxf(t12[i * 12 + k] * scsh[k] + scsh[12 + k], 0.f);
        acc += a * w2[k * 64 + j];
    }
    h[t] = fmaxf(acc, 0.f);   // outer relu of first conv output
}

// ---------------- WMMA GEMMs (layer MLPs) ----------------
// ISA 7.12.2 operand maps for V_WMMA_F32_16X16X32_F16 (wave32):
//   A (16x32 f16): row = lane&15; half t -> k = ((t&8)?16:0) + ((lane>>4)<<3) + (t&7)
//   B (32x16 f16): col = lane&15; half t -> k = ((t&8)?16:0) + ((lane>>4)<<3) + (t&7)
//   C/D (16x16 f32): col = lane&15; vgpr r -> row = ((lane>>4)<<3) + r
__global__ void k_gemm1_wmma(const float* __restrict__ A, const float* __restrict__ W,
                             const float* __restrict__ bias, float* __restrict__ Hid,
                             float* __restrict__ stats) {
    int wave = (blockIdx.x * blockDim.x + threadIdx.x) >> 5;
    int lane = threadIdx.x & 31;
    if (wave >= NNODES / 16) return;
    int m0  = wave * 16;
    int row = m0 + (lane & 15);
    int hi  = lane >> 4;
    v16h a0, a1;
    #pragma unroll
    for (int t = 0; t < 16; ++t) {
        int k = ((t & 8) ? 16 : 0) + (hi << 3) + (t & 7);
        a0[t] = (_Float16)A[row * 64 + k];
        a1[t] = (_Float16)A[row * 64 + 32 + k];
    }
    int ncol = lane & 15;
    #pragma unroll
    for (int nt = 0; nt < 8; ++nt) {
        int n = nt * 16 + ncol;
        v16h b0, b1v;
        #pragma unroll
        for (int t = 0; t < 16; ++t) {
            int k = ((t & 8) ? 16 : 0) + (hi << 3) + (t & 7);
            b0[t]  = (_Float16)W[k * 128 + n];
            b1v[t] = (_Float16)W[(32 + k) * 128 + n];
        }
        v8f acc = {};
        acc = __builtin_amdgcn_wmma_f32_16x16x32_f16(false, a0, false, b0,  (short)0, acc, false, false);
        acc = __builtin_amdgcn_wmma_f32_16x16x32_f16(false, a1, false, b1v, (short)0, acc, false, false);
        float bn = bias[n];
        float s = 0.f, ss = 0.f;
        #pragma unroll
        for (int r = 0; r < 8; ++r) {
            int m = m0 + (hi << 3) + r;
            float v = acc[r] + bn;
            Hid[m * 128 + n] = v;
            s += v; ss += v * v;
        }
        atomicAdd(&stats[n], s);
        atomicAdd(&stats[128 + n], ss);
    }
}

// A = relu(bn(hidden)) applied on load; output adds residual into h (res+).
__global__ void k_gemm2_wmma(const float* __restrict__ Hid, const float* __restrict__ scsh,
                             const float* __restrict__ W, const float* __restrict__ bias,
                             float* __restrict__ h) {
    int wave = (blockIdx.x * blockDim.x + threadIdx.x) >> 5;
    int lane = threadIdx.x & 31;
    if (wave >= NNODES / 16) return;
    int m0  = wave * 16;
    int row = m0 + (lane & 15);
    int hi  = lane >> 4;
    v16h a[4];
    #pragma unroll
    for (int kc = 0; kc < 4; ++kc) {
        #pragma unroll
        for (int t = 0; t < 16; ++t) {
            int k = kc * 32 + ((t & 8) ? 16 : 0) + (hi << 3) + (t & 7);
            float v = Hid[row * 128 + k] * scsh[k] + scsh[128 + k];
            a[kc][t] = (_Float16)fmaxf(v, 0.f);
        }
    }
    int ncol = lane & 15;
    #pragma unroll
    for (int nt = 0; nt < 4; ++nt) {
        int n = nt * 16 + ncol;
        v8f acc = {};
        #pragma unroll
        for (int kc = 0; kc < 4; ++kc) {
            v16h b;
            #pragma unroll
            for (int t = 0; t < 16; ++t) {
                int k = kc * 32 + ((t & 8) ? 16 : 0) + (hi << 3) + (t & 7);
                b[t] = (_Float16)W[k * 64 + n];
            }
            acc = __builtin_amdgcn_wmma_f32_16x16x32_f16(false, a[kc], false, b, (short)0, acc, false, false);
        }
        float bn = bias[n];
        #pragma unroll
        for (int r = 0; r < 8; ++r) {
            int m = m0 + (hi << 3) + r;
            h[m * 64 + n] += acc[r] + bn;   // residual
        }
    }
}

// ---------------- pooling + head ----------------
__global__ void k_pool_max(const float* __restrict__ h, const int* __restrict__ batch,
                           float* __restrict__ pooled) {
    int t = blockIdx.x * blockDim.x + threadIdx.x;
    if (t >= NNODES * 64) return;
    int i = t >> 6, c = t & 63;
    float v = h[t];
    float* addr = &pooled[batch[i] * 64 + c];
    if (v >= 0.f) atomicMax((int*)addr, __float_as_int(v));
    else          atomicMin((unsigned int*)addr, __float_as_uint(v));
}

__global__ void k_final_mlp(const float* __restrict__ pooled, const float* __restrict__ w1,
                            const float* __restrict__ b1, const float* __restrict__ w2,
                            const float* __restrict__ b2, float* __restrict__ out) {
    __shared__ float g[64];
    __shared__ float y[64];
    int b = blockIdx.x, t = threadIdx.x;   // 80 threads
    if (t < 64) {
        float v = pooled[b * 64 + t];
        g[t] = (__float_as_uint(v) == 0xFF800000u) ? 0.f : v;  // isneginf -> 0
    }
    __syncthreads();
    if (t < 64) {
        float acc = b1[t];
        #pragma unroll 8
        for (int k = 0; k < 64; ++k) acc += g[k] * w1[k * 64 + t];
        y[t] = fmaxf(acc, 0.f);
    }
    __syncthreads();
    float acc = b2[t];
    #pragma unroll 8
    for (int k = 0; k < 64; ++k) acc += y[k] * w2[k * 80 + t];
    out[b * 80 + t] = acc;
}

// ---------------- host orchestration ----------------
extern "C" void kernel_launch(void* const* d_in, const int* in_sizes, int n_in,
                              void* d_out, int out_size, void* d_ws, size_t ws_size,
                              hipStream_t stream) {
    const float* x      = (const float*)d_in[0];
    const int*   ei     = (const int*)d_in[1];
    const int*   batch  = (const int*)d_in[2];
    const float* cin_w1 = (const float*)d_in[3];
    const float* cin_b1 = (const float*)d_in[4];
    const float* cin_g1 = (const float*)d_in[5];
    const float* cin_be1= (const float*)d_in[6];
    const float* cin_w2 = (const float*)d_in[7];
    const float* cin_b2 = (const float*)d_in[8];
    const float* L_w1   = (const float*)d_in[9];
    const float* L_b1   = (const float*)d_in[10];
    const float* L_g1   = (const float*)d_in[11];
    const float* L_be1  = (const float*)d_in[12];
    const float* L_w2   = (const float*)d_in[13];
    const float* L_b2   = (const float*)d_in[14];
    const float* mlp_w1 = (const float*)d_in[15];
    const float* mlp_b1 = (const float*)d_in[16];
    const float* mlp_w2 = (const float*)d_in[17];
    const float* mlp_b2 = (const float*)d_in[18];

    const int* src = ei;            // edge_index[0]
    const int* dst = ei + NEDGES;   // edge_index[1]

    // workspace layout (floats); regions aliased across phases
    float* W     = (float*)d_ws;
    float* h     = W;                 // [N,64]  persistent node state
    float* xin   = h    + N64;        // [N,64]  conv input (also cin out0 [N,6])
    float* bufC  = xin  + N64;        // [N,64]  mmax -> then GEMM1 A ("out")
    float* bufD  = bufC + N64;        // [N,64]  denom -> hidden low half
    float* bufE  = bufD + N64;        // [N,64]  agg   -> hidden high half
    float* hidden = bufD;             // [N,128] spans D+E (contiguous)
    float* t12   = bufE + N64;        // [N,12]
    float* stats = t12  + NNODES * 12;// [256]
    float* scsh  = stats + 256;       // [256]
    float* pooled = scsh + 256;       // [B,64]

    const int TB = 256;
    const int gE16 = cdiv(NEDGES * 16, TB);
    const int gE   = cdiv(NEDGES, TB);
    const int gN64 = cdiv(NNODES * 64, TB);
    const int gN   = cdiv(NNODES, TB);
    const int gWmma = cdiv((NNODES / 16) * 32, TB);

    // ===== cin conv (F=6, scalar MLP path) =====
    hipMemsetAsync(bufC, 0, (size_t)3 * N64 * sizeof(float), stream); // mmax/denom/agg
    k_edge_max6<<<gE, TB, 0, stream>>>(x, src, dst, (unsigned int*)bufC);
    k_edge_sum6<<<gE, TB, 0, stream>>>(x, src, dst, bufC, bufD);
    k_edge_agg6<<<gE, TB, 0, stream>>>(x, src, dst, bufC, bufD, bufE);
    k_add<<<cdiv(NNODES * 6, TB), TB, 0, stream>>>(bufE, x, xin, NNODES * 6); // out0 = agg + x
    hipMemsetAsync(stats, 0, 256 * sizeof(float), stream);
    k_cin_mlp1<<<gN, TB, 0, stream>>>(xin, cin_w1, cin_b1, t12, stats);
    k_bn_finalize<<<1, 32, 0, stream>>>(stats, cin_g1, cin_be1, scsh, 12, 1.f / NNODES);
    k_cin_mlp2<<<gN64, TB, 0, stream>>>(t12, scsh, cin_w2, cin_b2, h);

    // ===== 3 DeepGCN res+ layers (F=64, WMMA MLP path) =====
    for (int L = 0; L < 3; ++L) {
        const float* w1 = L_w1  + (size_t)L * 64 * 128;
        const float* b1 = L_b1  + (size_t)L * 128;
        const float* g1 = L_g1  + (size_t)L * 128;
        const float* be1= L_be1 + (size_t)L * 128;
        const float* w2 = L_w2  + (size_t)L * 128 * 64;
        const float* b2 = L_b2  + (size_t)L * 64;

        k_relu<<<gN64, TB, 0, stream>>>(h, xin, N64);                  // conv input = relu(h)
        hipMemsetAsync(bufC, 0, (size_t)3 * N64 * sizeof(float), stream);
        k_edge_max64<<<gE16, TB, 0, stream>>>((const float4*)xin, src, dst, (unsigned int*)bufC);
        k_edge_sum64<<<gE16, TB, 0, stream>>>((const float4*)xin, src, dst, (const float4*)bufC, bufD);
        k_edge_agg64<<<gE16, TB, 0, stream>>>((const float4*)xin, src, dst,
                                              (const float4*)bufC, (const float4*)bufD, bufE);
        k_add<<<gN64, TB, 0, stream>>>(bufE, xin, bufC, N64);          // out = agg + relu(h)
        hipMemsetAsync(stats, 0, 256 * sizeof(float), stream);
        k_gemm1_wmma<<<gWmma, TB, 0, stream>>>(bufC, w1, b1, hidden, stats);
        k_bn_finalize<<<1, 128, 0, stream>>>(stats, g1, be1, scsh, 128, 1.f / NNODES);
        k_gemm2_wmma<<<gWmma, TB, 0, stream>>>(hidden, scsh, w2, b2, h); // h += delta
    }

    // ===== global max pool + head =====
    k_fill_u32<<<cdiv(NGRAPH * 64, TB), TB, 0, stream>>>((unsigned int*)pooled, 0xFF800000u, NGRAPH * 64);
    k_pool_max<<<gN64, TB, 0, stream>>>(h, batch, pooled);
    k_final_mlp<<<NGRAPH, 80, 0, stream>>>(pooled, mlp_w1, mlp_b1, mlp_w2, mlp_b2, (float*)d_out);
}
